// BidirectionalAttentionV2_11991548690703
// MI455X (gfx1250) — compile-verified, matrix operand order checked by reading, not run on hardware
//
#include <hip/hip_runtime.h>
#include <hip/hip_fp16.h>

#define B_  8
#define NH_ 12
#define T_  1024
#define N_  256
#define QT  64      // query rows per workgroup (16 per wave)
#define ST  32      // key rows per inner iteration
#define NW  4       // waves per workgroup

typedef __attribute__((ext_vector_type(16))) _Float16 v16h;
typedef __attribute__((ext_vector_type(8)))  float    v8f;

union Frag {
    v16h     h;
    uint4    q[2];
    unsigned u[8];
};

union H4 {
    _Float16 h[4];
    uint2    u;
};

// Load a 16-bit A/B fragment from row-major LDS (ISA 7.12.2 layout).
// Per lane: elements cover halves K = [8*half, 8*half+8) and [16+8*half, +8)
// of the 32-wide K chunk -> two contiguous 16B runs -> two ds_load_b128.
__device__ __forceinline__ v16h load_frag(const _Float16* row_ptr, int half) {
    Frag f;
    const _Float16* p = row_ptr + (half << 3);
    f.q[0] = *(const uint4*)(p);        // halves +0 .. +7
    f.q[1] = *(const uint4*)(p + 16);   // halves +16 .. +23
    return f.h;
}

__global__ __launch_bounds__(128, 1)
void rope_attn_kernel(const float* __restrict__ Q, const float* __restrict__ V,
                      const float* __restrict__ freqs, float* __restrict__ out) {
    __shared__ _Float16 sQ[QT][N_];        // RoPE'd Q rows (A source)   32 KB
    __shared__ _Float16 sK[ST][N_];        // RoPE'd K tile (B source)   16 KB
    __shared__ _Float16 sVt[N_][ST];       // V tile, transposed         16 KB
    __shared__ _Float16 sP[NW][16][ST];    // softmax probs staging       4 KB

    const int tid  = threadIdx.x;
    const int wave = tid >> 5;
    const int lane = tid & 31;
    const int half = lane >> 4;
    const int ln   = lane & 15;

    const int q0 = blockIdx.x * QT;
    const size_t base = (size_t)blockIdx.y * T_ * N_;   // (b*NH + h) offset

    const float TWO_PI = 6.28318530717958647692f;
    const float scale  = 0.0625f;                        // 1/sqrt(256)

    // ---- stage RoPE'd Q tile (float4 global loads, uint2 LDS stores) ----
    for (int p = tid; p < QT * (N_ / 4); p += 128) {
        int row = p >> 6;                 // N_/4 == 64 quads per row
        int n   = (p & 63) * 4;
        int t   = q0 + row;
        float f0 = freqs[n], f1 = freqs[n + 2];
        float ph0 = (float)t * f0, ph1 = (float)t * f1;
        float a0 = (ph0 - floorf(ph0)) * TWO_PI;
        float a1 = (ph1 - floorf(ph1)) * TWO_PI;
        float c0v, s0v, c1v, s1v;
        __sincosf(a0, &s0v, &c0v);
        __sincosf(a1, &s1v, &c1v);
        const float4 qv = *(const float4*)(Q + base + (size_t)t * N_ + n);
        H4 w;
        w.h[0] = (_Float16)(qv.x * c0v - qv.y * s0v);
        w.h[1] = (_Float16)(qv.y * c0v + qv.x * s0v);
        w.h[2] = (_Float16)(qv.z * c1v - qv.w * s1v);
        w.h[3] = (_Float16)(qv.w * c1v + qv.z * s1v);
        *(uint2*)&sQ[row][n] = w.u;
    }

    // per-row online-softmax state; C-layout: VGPR r of this lane == row r + 8*half
    float mrow[8], lrow[8];
    v8f oacc[16];
    const v8f vzero = {};
#pragma unroll
    for (int r = 0; r < 8; ++r) { mrow[r] = -1e30f; lrow[r] = 0.0f; }
#pragma unroll
    for (int c = 0; c < 16; ++c) oacc[c] = vzero;

    for (int s0 = 0; s0 < T_; s0 += ST) {
        __syncthreads();   // previous iteration done with sK/sVt (also fences sQ staging)

        // ---- stage RoPE'd K tile + transposed V tile ----
        for (int p = tid; p < ST * (N_ / 4); p += 128) {
            int row = p >> 6;
            int n   = (p & 63) * 4;
            int t   = s0 + row;
            float f0 = freqs[n], f1 = freqs[n + 2];
            float ph0 = (float)t * f0, ph1 = (float)t * f1;
            float a0 = (ph0 - floorf(ph0)) * TWO_PI;
            float a1 = (ph1 - floorf(ph1)) * TWO_PI;
            float c0v, s0v, c1v, s1v;
            __sincosf(a0, &s0v, &c0v);
            __sincosf(a1, &s1v, &c1v);
            const float4 kv = *(const float4*)(Q + base + (size_t)t * N_ + n);
            H4 w;
            w.h[0] = (_Float16)(kv.x * c0v - kv.y * s0v);
            w.h[1] = (_Float16)(kv.y * c0v + kv.x * s0v);
            w.h[2] = (_Float16)(kv.z * c1v - kv.w * s1v);
            w.h[3] = (_Float16)(kv.w * c1v + kv.z * s1v);
            *(uint2*)&sK[row][n] = w.u;
            const float4 vv = *(const float4*)(V + base + (size_t)t * N_ + n);
            sVt[n][row]     = (_Float16)vv.x;
            sVt[n + 1][row] = (_Float16)vv.y;
            sVt[n + 2][row] = (_Float16)vv.z;
            sVt[n + 3][row] = (_Float16)vv.w;
            // prefetch next tile's K/V rows into cache while WMMAs run
            if (s0 + ST < T_) {
                __builtin_prefetch(Q + base + (size_t)(t + ST) * N_ + n, 0, 1);
                __builtin_prefetch(V + base + (size_t)(t + ST) * N_ + n, 0, 1);
            }
        }
        __syncthreads();

        // ---- S(16x32) = Qr(16x256) @ Kr^T : two 16x16 f32 accumulators ----
        v8f c0 = vzero, c1 = vzero;
        const _Float16* qrow = &sQ[wave * 16 + ln][0];
#pragma unroll
        for (int kc = 0; kc < 8; ++kc) {
            v16h a  = load_frag(qrow + kc * 32, half);               // A: row M=ln
            v16h b0 = load_frag(&sK[ln][0] + kc * 32, half);         // B: key col ln
            v16h b1 = load_frag(&sK[ln + 16][0] + kc * 32, half);    // B: key col ln+16
            c0 = __builtin_amdgcn_wmma_f32_16x16x32_f16(false, a, false, b0,
                                                        (short)0, c0, false, false);
            c1 = __builtin_amdgcn_wmma_f32_16x16x32_f16(false, a, false, b1,
                                                        (short)0, c1, false, false);
        }

        // ---- online softmax (row reductions across the 16-lane half) ----
        float alpha[8];
#pragma unroll
        for (int r = 0; r < 8; ++r) {
            float s0v = c0[r] * scale;
            float s1v = c1[r] * scale;
            float mx = fmaxf(s0v, s1v);
#pragma unroll
            for (int off = 1; off < 16; off <<= 1)
                mx = fmaxf(mx, __shfl_xor(mx, off, 32));
            float mnew = fmaxf(mrow[r], mx);
            float al   = __expf(mrow[r] - mnew);
            float p0   = __expf(s0v - mnew);
            float p1   = __expf(s1v - mnew);
            float rs   = p0 + p1;
#pragma unroll
            for (int off = 1; off < 16; off <<= 1)
                rs += __shfl_xor(rs, off, 32);
            lrow[r]  = lrow[r] * al + rs;
            mrow[r]  = mnew;
            alpha[r] = al;
            int m = r + half * 8;                 // C-layout row
            sP[wave][m][ln]      = (_Float16)p0;  // transpose P into A layout via LDS
            sP[wave][m][ln + 16] = (_Float16)p1;
        }
#pragma unroll
        for (int c = 0; c < 16; ++c)
#pragma unroll
            for (int r = 0; r < 8; ++r)
                oacc[c][r] *= alpha[r];

        // ---- O(16x256) += P(16x32) @ V(32x256) ----
        v16h pa = load_frag(&sP[wave][ln][0], half);                 // A: row M=ln, K=key
#pragma unroll
        for (int nc = 0; nc < 16; ++nc) {
            v16h bv = load_frag(&sVt[nc * 16 + ln][0], half);        // B: col n, K=key
            oacc[nc] = __builtin_amdgcn_wmma_f32_16x16x32_f16(false, pa, false, bv,
                                                              (short)0, oacc[nc], false, false);
        }
    }

    // ---- epilogue: out = O / l ----
    float* obase = out + base + (size_t)(q0 + wave * 16) * N_;
#pragma unroll
    for (int nc = 0; nc < 16; ++nc) {
#pragma unroll
        for (int r = 0; r < 8; ++r) {
            int m = r + half * 8;
            obase[(size_t)m * N_ + nc * 16 + ln] = oacc[nc][r] / lrow[r];
        }
    }
}

extern "C" void kernel_launch(void* const* d_in, const int* in_sizes, int n_in,
                              void* d_out, int out_size, void* d_ws, size_t ws_size,
                              hipStream_t stream) {
    (void)in_sizes; (void)n_in; (void)out_size; (void)d_ws; (void)ws_size;
    const float* Q     = (const float*)d_in[0];
    const float* V     = (const float*)d_in[1];
    const float* freqs = (const float*)d_in[2];
    float* out = (float*)d_out;

    dim3 grid(T_ / QT, B_ * NH_);
    rope_attn_kernel<<<grid, 128, 0, stream>>>(Q, V, freqs, out);
}
